// Goal_Policy_47321949667584
// MI455X (gfx1250) — matrix operand order repaired and checked by reading, not compiled, not run on hardware
//
#include <hip/hip_runtime.h>
#include <hip/hip_bf16.h>

typedef __attribute__((ext_vector_type(16))) __bf16 v16bf;
typedef __attribute__((ext_vector_type(8)))  __bf16 v8bf;
typedef __attribute__((ext_vector_type(8)))  float  v8f;
typedef __attribute__((ext_vector_type(4)))  float  v4f;

#define EMB   512
#define HID   128
#define ACT   18
#define MTILE 128   // batch rows per block

// XOR swizzle on 16-element (32B) column blocks of a 128-wide bf16 tile.
__device__ __forceinline__ int swz(int row, int col) {
  return row * 128 + ((((col >> 4) ^ row) & 7) << 4) + (col & 15);
}

struct AFrag { v4f f0, f1, f2, f3; };

__device__ __forceinline__ AFrag load_a_global(const float* __restrict__ xr,
                                               int kg, int hi8) {
  AFrag r;
  r.f0 = *(const v4f*)(xr + kg + hi8);
  r.f1 = *(const v4f*)(xr + kg + hi8 + 4);
  r.f2 = *(const v4f*)(xr + kg + 16 + hi8);
  r.f3 = *(const v4f*)(xr + kg + 16 + hi8 + 4);
  return r;
}

__device__ __forceinline__ v16bf pack_a(AFrag f) {
  v16bf a;
  #pragma unroll
  for (int j = 0; j < 4; ++j) {
    a[j]      = (__bf16)f.f0[j];
    a[4 + j]  = (__bf16)f.f1[j];
    a[8 + j]  = (__bf16)f.f2[j];
    a[12 + j] = (__bf16)f.f3[j];
  }
  return a;
}

__device__ __forceinline__ v16bf load_b(const __bf16* sW, int krow, int n) {
  const int ln = threadIdx.x & 31;
  const int row = n * 16 + (ln & 15);
  const int h8 = (ln >> 4) << 3;
  v8bf blo = *(const v8bf*)(&sW[swz(row, krow + h8)]);       // 16B contiguous, aligned
  v8bf bhi = *(const v8bf*)(&sW[swz(row, krow + 16 + h8)]);
  return __builtin_shufflevector(blo, bhi,
      0, 1, 2, 3, 4, 5, 6, 7, 8, 9, 10, 11, 12, 13, 14, 15);
}

__global__ __launch_bounds__(256, 1) void mlp_fused(
    const float* __restrict__ x,
    const float* __restrict__ W1, const float* __restrict__ b1,
    const float* __restrict__ W2, const float* __restrict__ b2,
    const float* __restrict__ W3, const float* __restrict__ b3,
    const float* __restrict__ W4, const float* __restrict__ b4,
    const float* __restrict__ W5, const float* __restrict__ b5,
    const float* __restrict__ W6, const float* __restrict__ b6,
    float* __restrict__ out)
{
  __shared__ __bf16 sW[HID * 128];    // staged W^T chunk (k-major), bf16, 32KB
  __shared__ __bf16 sH[MTILE * 128];  // activations, bf16, 32KB

  const int t   = threadIdx.x;
  const int l   = t & 31;          // lane in wave32
  const int w   = t >> 5;          // wave 0..7
  const int lo  = l & 15;
  const int hi8 = (l >> 4) << 3;   // 0 for lanes 0-15, 8 for lanes 16-31
  const int wr  = w * 16;          // wave's row base in the 128-row tile
  const int blockRow = blockIdx.x * MTILE;

  const v8f vzero = {0.f, 0.f, 0.f, 0.f, 0.f, 0.f, 0.f, 0.f};
  v8f acc[8];
  #pragma unroll
  for (int n = 0; n < 8; ++n) acc[n] = vzero;

  // ================= Layer 1: [B,512] x W1^T[512,128], K in 4 chunks =====
  const float* xr = x + (size_t)(blockRow + wr + lo) * EMB;
  #pragma unroll 1
  for (int ch = 0; ch < 4; ++ch) {
    const int kbase = ch * 128;
    __syncthreads();
    // stage sW[n][k] = bf16(W1[n][kbase+k]) for k<128, n<128 (coalesced b128)
    #pragma unroll
    for (int i = 0; i < 16; ++i) {
      int q  = t + i * 256;
      int k4 = (q & 31) << 2;
      int n  = q >> 5;
      v4f v = *(const v4f*)(W1 + (size_t)n * EMB + kbase + k4);
      sW[swz(n, k4 + 0)] = (__bf16)v[0];
      sW[swz(n, k4 + 1)] = (__bf16)v[1];
      sW[swz(n, k4 + 2)] = (__bf16)v[2];
      sW[swz(n, k4 + 3)] = (__bf16)v[3];
    }
    __syncthreads();

    AFrag fa = load_a_global(xr, kbase, hi8);   // A for ks=0 in flight
    #pragma unroll
    for (int ks = 0; ks < 4; ++ks) {
      v16bf a = pack_a(fa);
      if (ks < 3) fa = load_a_global(xr, kbase + (ks + 1) * 32, hi8); // prefetch
      const int krow = ks * 32;                 // B layout: lane = N, elements = K
      // 2-deep software pipeline of B fragments
      v16bf b0 = load_b(sW, krow, 0);
      v16bf b1 = load_b(sW, krow, 1);
      #pragma unroll
      for (int n = 0; n < 8; ++n) {
        v16bf b2 = b1;
        if (n < 6) b2 = load_b(sW, krow, n + 2);
        acc[n] = __builtin_amdgcn_wmma_f32_16x16x32_bf16(
            false, a, false, b0, (short)0, acc[n], false, false);
        b0 = b1; b1 = b2;
      }
    }
  }
  // bias + ReLU -> sH (wave-private rows; C layout: VGPR r -> M=r+hi8, N=lo)
  #pragma unroll
  for (int n = 0; n < 8; ++n) {
    int col = n * 16 + lo;
    float bv = b1[col];
    #pragma unroll
    for (int r = 0; r < 8; ++r) {
      float v = acc[n][r] + bv;
      sH[swz(wr + hi8 + r, col)] = (__bf16)fmaxf(v, 0.0f);
    }
  }

  // ================= Layers 2..5: [*,128] x W^T[128,128] =================
  const float* Wl[4] = {W2, W3, W4, W5};
  const float* bl[4] = {b2, b3, b4, b5};
  #pragma unroll 1
  for (int L = 0; L < 4; ++L) {
    const float* Wg = Wl[L];
    const float* bg = bl[L];
    __syncthreads();   // all waves done reading previous sW
    #pragma unroll
    for (int i = 0; i < 16; ++i) {
      int q  = t + i * 256;
      int k4 = (q & 31) << 2;
      int n  = q >> 5;
      v4f v = *(const v4f*)(Wg + n * HID + k4);
      sW[swz(n, k4 + 0)] = (__bf16)v[0];
      sW[swz(n, k4 + 1)] = (__bf16)v[1];
      sW[swz(n, k4 + 2)] = (__bf16)v[2];
      sW[swz(n, k4 + 3)] = (__bf16)v[3];
    }
    __syncthreads();
    #pragma unroll
    for (int n = 0; n < 8; ++n) acc[n] = vzero;

    // A fragment for ks=0 in flight
    v8bf alo = *(const v8bf*)(&sH[swz(wr + lo, 0 + hi8)]);
    v8bf ahi = *(const v8bf*)(&sH[swz(wr + lo, 16 + hi8)]);
    #pragma unroll
    for (int ks = 0; ks < 4; ++ks) {
      v16bf a = __builtin_shufflevector(alo, ahi,
          0, 1, 2, 3, 4, 5, 6, 7, 8, 9, 10, 11, 12, 13, 14, 15);
      if (ks < 3) {   // prefetch next k-step's A from LDS
        int k0n = (ks + 1) * 32;
        alo = *(const v8bf*)(&sH[swz(wr + lo, k0n + hi8)]);
        ahi = *(const v8bf*)(&sH[swz(wr + lo, k0n + 16 + hi8)]);
      }
      const int krow = ks * 32;
      v16bf b0 = load_b(sW, krow, 0);
      v16bf b1 = load_b(sW, krow, 1);
      #pragma unroll
      for (int n = 0; n < 8; ++n) {
        v16bf b2 = b1;
        if (n < 6) b2 = load_b(sW, krow, n + 2);
        acc[n] = __builtin_amdgcn_wmma_f32_16x16x32_bf16(
            false, a, false, b0, (short)0, acc[n], false, false);
        b0 = b1; b1 = b2;
      }
    }
    // bias + ReLU back into same rows (DS ops are in-order within a wave)
    #pragma unroll
    for (int n = 0; n < 8; ++n) {
      int col = n * 16 + lo;
      float bv = bg[col];
      #pragma unroll
      for (int r = 0; r < 8; ++r) {
        float v = acc[n][r] + bv;
        sH[swz(wr + hi8 + r, col)] = (__bf16)fmaxf(v, 0.0f);
      }
    }
  }

  // ================= Layer 6 (N=18) + softmax, VALU ======================
  if (l < 16) {
    const int row = wr + l;                 // lane owns one batch row
    float sc[ACT];
    #pragma unroll
    for (int a = 0; a < ACT; ++a) sc[a] = b6[a];
    #pragma unroll 1
    for (int blk = 0; blk < 8; ++blk) {
      float hb[16];
      const __bf16* hp = &sH[swz(row, blk * 16)];
      #pragma unroll
      for (int j = 0; j < 16; ++j) hb[j] = (float)hp[j];
      #pragma unroll
      for (int a = 0; a < ACT; ++a) {
        const float* wp = W6 + a * HID + blk * 16;  // uniform -> scalar loads
        float s = sc[a];
        #pragma unroll
        for (int j = 0; j < 16; ++j) s = fmaf(hb[j], wp[j], s);
        sc[a] = s;
      }
    }
    float m = sc[0];
    #pragma unroll
    for (int a = 1; a < ACT; ++a) m = fmaxf(m, sc[a]);
    float sum = 0.f;
    #pragma unroll
    for (int a = 0; a < ACT; ++a) { sc[a] = __expf(sc[a] - m); sum += sc[a]; }
    float inv = 1.0f / sum;
    float* op = out + (size_t)(blockRow + row) * ACT;
    #pragma unroll
    for (int a = 0; a < ACT; ++a) op[a] = sc[a] * inv;
  }
}

extern "C" void kernel_launch(void* const* d_in, const int* in_sizes, int n_in,
                              void* d_out, int out_size, void* d_ws, size_t ws_size,
                              hipStream_t stream) {
  const float* x  = (const float*)d_in[0];
  const float* W1 = (const float*)d_in[1];  const float* b1 = (const float*)d_in[2];
  const float* W2 = (const float*)d_in[3];  const float* b2 = (const float*)d_in[4];
  const float* W3 = (const float*)d_in[5];  const float* b3 = (const float*)d_in[6];
  const float* W4 = (const float*)d_in[7];  const float* b4 = (const float*)d_in[8];
  const float* W5 = (const float*)d_in[9];  const float* b5 = (const float*)d_in[10];
  const float* W6 = (const float*)d_in[11]; const float* b6 = (const float*)d_in[12];
  float* out = (float*)d_out;
  int batch = in_sizes[0] / EMB;
  dim3 grid(batch / MTILE);
  mlp_fused<<<grid, 256, 0, stream>>>(x, W1, b1, W2, b2, W3, b3, W4, b4,
                                      W5, b5, W6, b6, out);
}
